// NeuralGPU_69621419868185
// MI455X (gfx1250) — matrix operand order repaired and checked by reading, not compiled, not run on hardware
//
#include <hip/hip_runtime.h>

typedef __attribute__((ext_vector_type(16))) _Float16 v16h;
typedef __attribute__((ext_vector_type(8)))  _Float16 v8h;
typedef __attribute__((ext_vector_type(8)))  float    v8f;

#define B_     8
#define S_     4096
#define SP_    (S_ + 2)          // padded time dim: row 0 and row SP_-1 are zero
#define IN_    64
#define HID_   512
#define KW_    3
#define KC_    (KW_ * HID_)      // 1536
#define STEPS_ 12

#define KS_     128              // K-values per LDS weight stage (never crosses a tap)
#define NSTAGE_ (KC_ / KS_)      // 12
#define WPAD_   (KS_ + 8)        // padded LDS row: 136 halves = 272B (kills bank aliasing)
#define OPAD_   72               // padded out-stage row: 72 halves = 144B

static __device__ __forceinline__ v16h cat8(v8h lo, v8h hi) {
  return __builtin_shufflevector(lo, hi, 0,1,2,3,4,5,6,7,8,9,10,11,12,13,14,15);
}

#define WMMA_F16(a, b, c) \
  __builtin_amdgcn_wmma_f32_16x16x32_f16(false, (a), false, (b), (short)0, (c), false, false)

// ---------------------------------------------------------------------------
// Input projection: h[b,s,:] = x[b,s,:] @ W_in + b_in   (M=B*S, K=64, N=512)
// ---------------------------------------------------------------------------
__global__ void __launch_bounds__(256)
k_inproj(const _Float16* __restrict__ xh, const _Float16* __restrict__ WinT,
         const float* __restrict__ bin, _Float16* __restrict__ hout) {
  const int lane = threadIdx.x & 31, wave = threadIdx.x >> 5;
  const int lr = lane & 15, lhi = lane >> 4;
  const int pos0 = (blockIdx.x * 8 + wave) * 16;
  const int b = pos0 / S_, s0 = pos0 % S_;
  const int n0 = blockIdx.y * 64;

  const _Float16* arow = xh  + (size_t)(pos0 + lr) * IN_ + lhi * 8;
  const _Float16* brow = WinT + (size_t)(n0 + lr) * IN_ + lhi * 8;

  v8f zero = {0.f,0.f,0.f,0.f,0.f,0.f,0.f,0.f};
  v8f acc[4] = {zero, zero, zero, zero};

  for (int kc = 0; kc < IN_; kc += 32) {
    v16h a = cat8(*(const v8h*)(arow + kc), *(const v8h*)(arow + kc + 16));
#pragma unroll
    for (int t = 0; t < 4; ++t) {
      const _Float16* bp = brow + (size_t)t * 16 * IN_ + kc;
      v16h bf = cat8(*(const v8h*)bp, *(const v8h*)(bp + 16));
      acc[t] = WMMA_F16(a, bf, acc[t]);
    }
  }

  _Float16* obase = hout + ((size_t)b * SP_ + (size_t)(s0 + 1)) * HID_;
#pragma unroll
  for (int t = 0; t < 4; ++t) {
    const int n = n0 + t * 16 + lr;
    const float bias = bin[n];
#pragma unroll
    for (int j = 0; j < 8; ++j) {
      const int r = j + lhi * 8;
      obase[(size_t)r * HID_ + n] = (_Float16)(acc[t][j] + bias);
    }
  }
}

// ---------------------------------------------------------------------------
// Conv step as GEMM (M=B*S, N=512, K=1536 = tap*512+cin), padded rows give
// zero boundaries. Weights double-buffered through LDS, shared by 8 waves.
// Block: 256 positions x 64 couts. Wave: 32 positions x 64 couts
// (2 M-tiles x 4 N-tiles = 8 accumulators; each B fragment feeds 2 WMMAs).
// ---------------------------------------------------------------------------
__global__ void __launch_bounds__(256)
k_conv(const _Float16* __restrict__ hin, const _Float16* __restrict__ Wt,
       const float* __restrict__ bconv, _Float16* __restrict__ hout) {
  __shared__ _Float16 sW[2][64][WPAD_];       // 2 x 64 couts x 128 K (padded) = 34 KB
  __shared__ _Float16 sO[8][16][OPAD_];       // per-wave 16x64 out tile (padded) = 18 KB

  const int tid  = threadIdx.x;
  const int lane = tid & 31, wave = tid >> 5;
  const int lr = lane & 15, lhi = lane >> 4;
  const int pos0 = (blockIdx.x * 8 + wave) * 32;       // 32 positions per wave
  const int b = pos0 / S_, s0 = pos0 % S_;
  const int n0 = blockIdx.y * 64;

  // A: output position s reads padded rows s, s+1, s+2 (tap 0..2)
  const _Float16* arow0 = hin + ((size_t)b * SP_ + (size_t)(s0 + lr)) * HID_ + lhi * 8;
  // cooperative weight stage: thread -> (row = tid/4, 32 contiguous K at (tid%4)*32)
  const int wrow = tid >> 2;
  const int wkof = (tid & 3) * 32;
  const _Float16* wsrc = Wt + (size_t)(n0 + wrow) * KC_ + wkof;

  v8f zero = {0.f,0.f,0.f,0.f,0.f,0.f,0.f,0.f};
  v8f acc[2][4] = {{zero, zero, zero, zero}, {zero, zero, zero, zero}};

  // prologue: stage 0
  {
    v8h* dst = (v8h*)&sW[0][wrow][wkof];
#pragma unroll
    for (int i = 0; i < 4; ++i) dst[i] = *(const v8h*)(wsrc + i * 8);
  }
  __syncthreads();

  for (int st = 0; st < NSTAGE_; ++st) {
    const int cur = st & 1;
    // prefetch next stage into the other buffer (overlaps with WMMAs below)
    if (st + 1 < NSTAGE_) {
      const _Float16* gp = wsrc + (size_t)(st + 1) * KS_;
      v8h* dst = (v8h*)&sW[cur ^ 1][wrow][wkof];
#pragma unroll
      for (int i = 0; i < 4; ++i) dst[i] = *(const v8h*)(gp + i * 8);
    }

    const int kg0 = st * KS_;
    const int tap = kg0 >> 9;          // KS_=128 never crosses a tap boundary
    const int cin0 = kg0 & (HID_ - 1);
    const _Float16* ar0 = arow0 + (size_t)tap * HID_ + cin0;
    const _Float16* ar1 = ar0 + (size_t)16 * HID_;      // second M tile: +16 positions

#pragma unroll
    for (int c = 0; c < KS_ / 32; ++c) {
      v16h a0 = cat8(*(const v8h*)(ar0 + c * 32), *(const v8h*)(ar0 + c * 32 + 16));
      v16h a1 = cat8(*(const v8h*)(ar1 + c * 32), *(const v8h*)(ar1 + c * 32 + 16));
#pragma unroll
      for (int t = 0; t < 4; ++t) {
        const _Float16* bp = &sW[cur][t * 16 + lr][c * 32 + lhi * 8];
        v16h bf = cat8(*(const v8h*)bp, *(const v8h*)(bp + 16));
        acc[0][t] = WMMA_F16(a0, bf, acc[0][t]);
        acc[1][t] = WMMA_F16(a1, bf, acc[1][t]);
      }
    }
    __syncthreads();
  }

  // epilogue: bias + relu, transpose through per-wave LDS patch, coalesced store.
  // Two sequential passes over the same per-wave patch (wave LDS ops are in-order).
  _Float16* obase = hout + ((size_t)b * SP_ + (size_t)(s0 + 1)) * HID_;
#pragma unroll
  for (int m = 0; m < 2; ++m) {
#pragma unroll
    for (int t = 0; t < 4; ++t) {
      const float bias = bconv[n0 + t * 16 + lr];
#pragma unroll
      for (int j = 0; j < 8; ++j) {
        float v = acc[m][t][j] + bias;
        sO[wave][j + lhi * 8][t * 16 + lr] = (_Float16)(v > 0.f ? v : 0.f);
      }
    }
    // each lane streams out 32 contiguous halves (4 x b128)
    const int r = lane >> 1;
    const int cg = (lane & 1) * 32;
    const v8h* srcp = (const v8h*)&sO[wave][r][cg];
    _Float16* gout = obase + (size_t)(m * 16 + r) * HID_ + n0 + cg;
#pragma unroll
    for (int i = 0; i < 4; ++i) *(v8h*)(gout + i * 8) = srcp[i];
  }
}

// ---------------------------------------------------------------------------
// Output projection: out[b,s,:] = h[b,s,:] @ W_out + b_out (M=B*S, K=512, N=64)
// ---------------------------------------------------------------------------
__global__ void __launch_bounds__(256)
k_outproj(const _Float16* __restrict__ hin, const _Float16* __restrict__ WoT,
          const float* __restrict__ bout, float* __restrict__ out) {
  const int lane = threadIdx.x & 31, wave = threadIdx.x >> 5;
  const int lr = lane & 15, lhi = lane >> 4;
  const int pos0 = (blockIdx.x * 8 + wave) * 16;
  const int b = pos0 / S_, s0 = pos0 % S_;

  const _Float16* arow = hin + ((size_t)b * SP_ + (size_t)(s0 + lr + 1)) * HID_ + lhi * 8;
  const _Float16* brow = WoT + (size_t)lr * HID_ + lhi * 8;

  v8f zero = {0.f,0.f,0.f,0.f,0.f,0.f,0.f,0.f};
  v8f acc[4] = {zero, zero, zero, zero};

  for (int kc = 0; kc < HID_; kc += 32) {
    v16h a = cat8(*(const v8h*)(arow + kc), *(const v8h*)(arow + kc + 16));
#pragma unroll
    for (int t = 0; t < 4; ++t) {
      const _Float16* bp = brow + (size_t)t * 16 * HID_ + kc;
      v16h bf = cat8(*(const v8h*)bp, *(const v8h*)(bp + 16));
      acc[t] = WMMA_F16(a, bf, acc[t]);
    }
  }

#pragma unroll
  for (int t = 0; t < 4; ++t) {
    const int n = t * 16 + lr;
    const float bias = bout[n];
#pragma unroll
    for (int j = 0; j < 8; ++j) {
      const int r = j + lhi * 8;
      out[(size_t)(pos0 + r) * IN_ + n] = acc[t][j] + bias;
    }
  }
}

// ------------------------- data prep kernels -------------------------------
__global__ void k_f32_to_f16(const float* __restrict__ src, _Float16* __restrict__ dst, int n) {
  int i = blockIdx.x * blockDim.x + threadIdx.x;
  if (i < n) dst[i] = (_Float16)src[i];
}

// WinT[n][k] = W_in[k][n]   (W_in: 64x512)
__global__ void k_prep_win(const float* __restrict__ Win, _Float16* __restrict__ WinT) {
  int i = blockIdx.x * blockDim.x + threadIdx.x;
  if (i < HID_ * IN_) {
    int n = i / IN_, k = i % IN_;
    WinT[i] = (_Float16)Win[(size_t)k * HID_ + n];
  }
}

// Wt[cout][tap*512+cin] = W_conv[cout][cin][tap]   (W_conv: 512x512x3)
__global__ void k_prep_wconv(const float* __restrict__ Wc, _Float16* __restrict__ Wt) {
  int i = blockIdx.x * blockDim.x + threadIdx.x;
  if (i < HID_ * KC_) {
    int cout = i / KC_, r = i % KC_;
    int tap = r / HID_, cin = r % HID_;
    Wt[i] = (_Float16)Wc[((size_t)cout * HID_ + cin) * KW_ + tap];
  }
}

// WoT[n][k] = W_out[k][n]   (W_out: 512x64)
__global__ void k_prep_wout(const float* __restrict__ Wo, _Float16* __restrict__ WoT) {
  int i = blockIdx.x * blockDim.x + threadIdx.x;
  if (i < IN_ * HID_) {
    int n = i / HID_, k = i % HID_;
    WoT[i] = (_Float16)Wo[(size_t)k * IN_ + n];
  }
}

// zero the two pad rows of both ping/pong buffers (every launch; deterministic)
__global__ void k_zero_pads(_Float16* __restrict__ hA, _Float16* __restrict__ hB) {
  int i = blockIdx.x * blockDim.x + threadIdx.x;
  if (i < B_ * 2 * HID_) {
    int b = i / (2 * HID_);
    int r = i % (2 * HID_);
    int side = r / HID_, c = r % HID_;
    size_t row = (size_t)b * SP_ + (side ? (SP_ - 1) : 0);
    hA[row * HID_ + c] = (_Float16)0.f;
    hB[row * HID_ + c] = (_Float16)0.f;
  }
}

// ---------------------------------------------------------------------------
extern "C" void kernel_launch(void* const* d_in, const int* in_sizes, int n_in,
                              void* d_out, int out_size, void* d_ws, size_t ws_size,
                              hipStream_t stream) {
  (void)in_sizes; (void)n_in; (void)out_size; (void)ws_size;
  const float* x   = (const float*)d_in[0];
  const float* Win = (const float*)d_in[1];
  const float* bin = (const float*)d_in[2];
  const float* Wc  = (const float*)d_in[3];
  const float* bc  = (const float*)d_in[4];
  const float* Wo  = (const float*)d_in[5];
  const float* bo  = (const float*)d_in[6];
  float* out = (float*)d_out;

  char* ws = (char*)d_ws;
  auto al = [](size_t v) { return (v + 255) & ~(size_t)255; };
  constexpr size_t NX = (size_t)B_ * S_ * IN_;        // 2,097,152
  constexpr size_t NH = (size_t)B_ * SP_ * HID_;      // 16,785,408 (padded)

  size_t off = 0;
  _Float16* xh   = (_Float16*)(ws + off); off = al(off + NX * 2);
  _Float16* hA   = (_Float16*)(ws + off); off = al(off + NH * 2);
  _Float16* hB   = (_Float16*)(ws + off); off = al(off + NH * 2);
  _Float16* Wt   = (_Float16*)(ws + off); off = al(off + (size_t)HID_ * KC_ * 2);
  _Float16* WinT = (_Float16*)(ws + off); off = al(off + (size_t)HID_ * IN_ * 2);
  _Float16* WoT  = (_Float16*)(ws + off); off = al(off + (size_t)IN_ * HID_ * 2);

  k_f32_to_f16<<<dim3((unsigned)((NX + 255) / 256)), 256, 0, stream>>>(x, xh, (int)NX);
  k_prep_win  <<<dim3((HID_ * IN_ + 255) / 256), 256, 0, stream>>>(Win, WinT);
  k_prep_wconv<<<dim3((HID_ * KC_ + 255) / 256), 256, 0, stream>>>(Wc, Wt);
  k_prep_wout <<<dim3((IN_ * HID_ + 255) / 256), 256, 0, stream>>>(Wo, WoT);
  k_zero_pads <<<dim3((B_ * 2 * HID_ + 255) / 256), 256, 0, stream>>>(hA, hB);

  dim3 gp(B_ * S_ / 128, HID_ / 64);    // projections: 128 pos x 64 couts per block
  k_inproj<<<gp, 256, 0, stream>>>(xh, WinT, bin, hA);

  dim3 gc(B_ * S_ / 256, HID_ / 64);    // conv: 256 pos x 64 couts per block
  _Float16* cur = hA; _Float16* nxt = hB;
  for (int step = 0; step < STEPS_; ++step) {
    k_conv<<<gc, 256, 0, stream>>>(cur, Wt, bc, nxt);
    _Float16* tmp = cur; cur = nxt; nxt = tmp;
  }

  k_outproj<<<dim3(B_ * S_ / 128, 1), 256, 0, stream>>>(cur, WoT, bo, out);
}
